// MyRelationalConv_18150531793634
// MI455X (gfx1250) — matrix operand order repaired and checked by reading, not compiled, not run on hardware
//
#include <hip/hip_runtime.h>

typedef __attribute__((ext_vector_type(16))) _Float16 v16h;
typedef __attribute__((ext_vector_type(8)))  _Float16 v8h;
typedef __attribute__((ext_vector_type(4)))  _Float16 v4h;
typedef __attribute__((ext_vector_type(8)))  float    v8f;
typedef __attribute__((ext_vector_type(4)))  float    f32x4;

#define N_NODES 8192
#define H_DIM   256
#define R_REL   3

#define BM 64          // rows per workgroup tile
#define KB 64          // K-slice per LDS stage (2 WMMA k-steps)
#define PA 72          // LDS pitch (halves) for A rows   (72*2=144B, conflict-free)
#define PB 72          // LDS pitch (halves) for B columns

// ---------------------------------------------------------------------------
// Fragment loaders (wave32) using this code's contiguous k order k = 16*hi + s for both A and B.
// A (16x32 f16): lane l<16 holds row M=l, K = 0..15; lanes 16..31 hold
//                row M=l-16, K = 16..31.  Two 16B LDS reads per lane.
// B (32x16 f16): lane l<16 holds col N=l, K=0..15; lanes 16..31 col N=l-16,
//                K=16..31.  Two 16B LDS reads per lane (column stored contiguously).
// ---------------------------------------------------------------------------
__device__ __forceinline__ v16h load_a_frag(const _Float16* __restrict__ as,
                                            int rowbase, int k2, int l, int hi) {
    const _Float16* p = as + (rowbase + l) * PA + k2 + hi * 16;
    v8h lo = *(const v8h*)(p);        // K = k2 + hi*16 .. +7
    v8h hh = *(const v8h*)(p + 8);    // K = k2 + hi*16 + 8 .. +15
    return __builtin_shufflevector(lo, hh, 0,1,2,3,4,5,6,7,8,9,10,11,12,13,14,15);
}

__device__ __forceinline__ v16h load_b_frag(const _Float16* __restrict__ bt,
                                            int colbase, int k2, int l, int hi) {
    const _Float16* p = bt + (colbase + l) * PB + k2 + hi * 16;
    v8h lo = *(const v8h*)(p);        // K = k2 + hi*16 .. +7
    v8h hh = *(const v8h*)(p + 8);    // K = k2 + hi*16 + 8 .. +15
    return __builtin_shufflevector(lo, hh, 0,1,2,3,4,5,6,7,8,9,10,11,12,13,14,15);
}

// 8 waves arranged 2(M) x 4(N): wave tile 32x64 => 2x4 WMMA accumulators.
__device__ __forceinline__ void mma_tile(const _Float16* __restrict__ as,
                                         const _Float16* __restrict__ bt,
                                         v8f acc[2][4], int wm, int wn,
                                         int l, int hi) {
#pragma unroll
    for (int k2 = 0; k2 < KB; k2 += 32) {
        v16h af[2];
        v16h bf[4];
#pragma unroll
        for (int mi = 0; mi < 2; ++mi)
            af[mi] = load_a_frag(as, wm * 32 + mi * 16, k2, l, hi);
#pragma unroll
        for (int ni = 0; ni < 4; ++ni)
            bf[ni] = load_b_frag(bt, wn * 64 + ni * 16, k2, l, hi);
#pragma unroll
        for (int mi = 0; mi < 2; ++mi)
#pragma unroll
            for (int ni = 0; ni < 4; ++ni)
                acc[mi][ni] = __builtin_amdgcn_wmma_f32_16x16x32_f16(
                    false, af[mi], false, bf[ni], (short)0, acc[mi][ni],
                    false, false);
    }
}

// Issue next A-tile global loads (fp32, streaming) into registers.
__device__ __forceinline__ void load_a_regs(f32x4 areg[4],
                                            const float* __restrict__ A,
                                            long m0, int kk, int t) {
    const int c4 = t & 15;
    const int r0 = t >> 4;
#pragma unroll
    for (int i = 0; i < 4; ++i) {
        const int row = r0 + i * 16;
        areg[i] = __builtin_nontemporal_load(
            (const f32x4*)(A + (m0 + row) * (long)N_NODES + kk + c4 * 4));
    }
}

// Convert staged registers to f16 and write the LDS A tile.
__device__ __forceinline__ void store_a_lds(_Float16* __restrict__ as,
                                            const f32x4 areg[4], int t) {
    const int c4 = t & 15;
    const int r0 = t >> 4;
#pragma unroll
    for (int i = 0; i < 4; ++i) {
        const int row = r0 + i * 16;
        v4h h;
        h[0] = (_Float16)areg[i][0]; h[1] = (_Float16)areg[i][1];
        h[2] = (_Float16)areg[i][2]; h[3] = (_Float16)areg[i][3];
        *(v4h*)(as + row * PA + c4 * 4) = h;
    }
}

// Async Global->LDS copy of one 128B xwt column slice (f16, no conversion).
// INST_OFFSET applies to both LDS and global addresses, and both sides are
// contiguous per thread, so one base pair + literal offsets covers the copy.
__device__ __forceinline__ void async_copy_b(_Float16* __restrict__ dst,
                                             const _Float16* __restrict__ src) {
    const unsigned ldsoff = (unsigned)(uintptr_t)dst;   // low 32 bits = LDS byte addr
#define ASYNC_B128(OFF) \
    asm volatile("global_load_async_to_lds_b128 %0, %1, off offset:" #OFF \
                 :: "v"(ldsoff), "v"(src) : "memory")
    ASYNC_B128(0);  ASYNC_B128(16); ASYNC_B128(32); ASYNC_B128(48);
    ASYNC_B128(64); ASYNC_B128(80); ASYNC_B128(96); ASYNC_B128(112);
#undef ASYNC_B128
}

__device__ __forceinline__ void wait_async0() {
    asm volatile("s_wait_asynccnt 0x0" ::: "memory");
}

// ---------------------------------------------------------------------------
// Kernel 1: xwt[r][c][n] = (x @ W[r])[n][c] in f16 (B stored pre-transposed
// so the hot kernel's B fragments are contiguous LDS reads).
// grid = (N/BM, R), block = 256
// ---------------------------------------------------------------------------
__global__ __launch_bounds__(256) void rgcn_xw_kernel(
        const float* __restrict__ x, const float* __restrict__ W,
        _Float16* __restrict__ xwt) {
    __shared__ __align__(16) _Float16 as[BM * PA];
    __shared__ __align__(16) _Float16 bt[H_DIM * PB];

    const int t = threadIdx.x;
    const int nb = blockIdx.x, r = blockIdx.y;
    const long n0 = (long)nb * BM;
    const int wave = t >> 5, lane = t & 31, l = lane & 15, hi = lane >> 4;
    const int wm = wave >> 2, wn = wave & 3;

    const float* __restrict__ Wr = W + (size_t)r * H_DIM * H_DIM;
    v8f acc[2][4] = {};

    for (int kk = 0; kk < H_DIM; kk += KB) {
        // stage x[n0..n0+63][kk..kk+63] (leading dim H_DIM) into LDS as f16
        {
            const int c4 = t & 15;
            const int r0 = t >> 4;
            f32x4 areg[4];
#pragma unroll
            for (int i = 0; i < 4; ++i) {
                const int row = r0 + i * 16;
                areg[i] = *(const f32x4*)(x + (n0 + row) * (long)H_DIM + kk + c4 * 4);
            }
            store_a_lds(as, areg, t);
        }
        // stage W[r][kk..kk+63][0..255] transposed into bt[col][k] as f16
        {
            const int c0 = (t & 63) * 4;
            const int rb = t >> 6;
#pragma unroll
            for (int i = 0; i < 16; ++i) {
                const int row = rb + i * 4;
                f32x4 v = *(const f32x4*)(Wr + (size_t)(kk + row) * H_DIM + c0);
#pragma unroll
                for (int e = 0; e < 4; ++e)
                    bt[(c0 + e) * PB + row] = (_Float16)v[e];
            }
        }
        __syncthreads();
        mma_tile(as, bt, acc, wm, wn, l, hi);
        __syncthreads();
    }

    // Store D transposed as f16: xwt[r][col][n]
    _Float16* __restrict__ Xr = xwt + (size_t)r * H_DIM * N_NODES;
#pragma unroll
    for (int mi = 0; mi < 2; ++mi)
#pragma unroll
        for (int ni = 0; ni < 4; ++ni)
#pragma unroll
            for (int j = 0; j < 8; ++j) {
                const int row = wm * 32 + mi * 16 + hi * 8 + j;
                const int col = wn * 64 + ni * 16 + l;
                Xr[(size_t)col * N_NODES + n0 + row] = (_Float16)acc[mi][ni][j];
            }
}

// ---------------------------------------------------------------------------
// Kernel 2 (hot): partial[r] = adjs[r] @ xw[r].  adjs streamed fp32->f16 with
// double-buffered LDS; xwt tile copied via async Global->LDS (ASYNCcnt).
// grid = (N/BM, R), block = 256.
// ---------------------------------------------------------------------------
__global__ __launch_bounds__(256) void rgcn_spmm_kernel(
        const float* __restrict__ adjs, const _Float16* __restrict__ xwt,
        float* __restrict__ partial) {
    __shared__ __align__(16) _Float16 as[2][BM * PA];
    __shared__ __align__(16) _Float16 bt[2][H_DIM * PB];

    const int t = threadIdx.x;
    const int mb = blockIdx.x, r = blockIdx.y;
    const long m0 = (long)mb * BM;
    const int wave = t >> 5, lane = t & 31, l = lane & 15, hi = lane >> 4;
    const int wm = wave >> 2, wn = wave & 3;

    const float*    __restrict__ A  = adjs + (size_t)r * N_NODES * N_NODES;
    const _Float16* __restrict__ Bg = xwt  + (size_t)r * H_DIM * N_NODES;
    v8f acc[2][4] = {};

    // Prologue: stage tile 0 into buffer 0
    {
        f32x4 areg[4];
        load_a_regs(areg, A, m0, 0, t);
        async_copy_b(&bt[0][t * PB], Bg + (size_t)t * N_NODES);
        store_a_lds(as[0], areg, t);
        wait_async0();
        __syncthreads();
    }

    int cur = 0;
    for (int kk = 0; kk < N_NODES; kk += KB) {
        const int nxt = cur ^ 1;
        const bool more = (kk + KB) < N_NODES;
        f32x4 areg[4];
        if (more) {
            // Issue next tile's loads before the math so they overlap WMMAs.
            load_a_regs(areg, A, m0, kk + KB, t);
            async_copy_b(&bt[nxt][t * PB], Bg + (size_t)t * N_NODES + kk + KB);
        }
        mma_tile(as[cur], bt[cur], acc, wm, wn, l, hi);
        if (more) {
            store_a_lds(as[nxt], areg, t);   // waits loadcnt here, after WMMAs
            wait_async0();                   // async B writes visible pre-barrier
        }
        __syncthreads();
        cur = nxt;
    }

    float* __restrict__ P = partial + (size_t)r * N_NODES * H_DIM;
#pragma unroll
    for (int mi = 0; mi < 2; ++mi)
#pragma unroll
        for (int ni = 0; ni < 4; ++ni)
#pragma unroll
            for (int j = 0; j < 8; ++j) {
                const int row = wm * 32 + mi * 16 + hi * 8 + j;
                const int col = wn * 64 + ni * 16 + l;
                P[(m0 + row) * H_DIM + col] = acc[mi][ni][j];
            }
}

// ---------------------------------------------------------------------------
// Kernel 3: out = partial[0] + partial[1] + partial[2]
// ---------------------------------------------------------------------------
__global__ __launch_bounds__(256) void rgcn_reduce_kernel(
        const float* __restrict__ partial, float* __restrict__ out) {
    const size_t idx = (size_t)blockIdx.x * 256 + threadIdx.x;
    const size_t plane = (size_t)N_NODES * H_DIM / 4;
    const f32x4* p = (const f32x4*)partial;
    f32x4 v = p[idx];
    v += p[idx + plane];
    v += p[idx + 2 * plane];
    ((f32x4*)out)[idx] = v;
}

extern "C" void kernel_launch(void* const* d_in, const int* in_sizes, int n_in,
                              void* d_out, int out_size, void* d_ws, size_t ws_size,
                              hipStream_t stream) {
    (void)in_sizes; (void)n_in; (void)out_size; (void)ws_size;
    const float* x    = (const float*)d_in[0];
    const float* adjs = (const float*)d_in[1];
    const float* W    = (const float*)d_in[2];
    float* out = (float*)d_out;

    _Float16* xwt = (_Float16*)d_ws;                               // 12 MB
    float* partial = (float*)((char*)d_ws + (size_t)12582912);     // 24 MB

    dim3 blk(256, 1, 1);
    dim3 grid1(N_NODES / BM, R_REL, 1);   // 128 x 3
    rgcn_xw_kernel<<<grid1, blk, 0, stream>>>(x, W, xwt);
    rgcn_spmm_kernel<<<grid1, blk, 0, stream>>>(adjs, xwt, partial);

    dim3 grid3((N_NODES * H_DIM / 4) / 256, 1, 1);  // 2048
    rgcn_reduce_kernel<<<grid3, blk, 0, stream>>>(partial, out);
}
